// sLSTMCell_47175920779410
// MI455X (gfx1250) — matrix operand order repaired and checked by reading, not compiled, not run on hardware
//
#include <hip/hip_runtime.h>

// ---------------------------------------------------------------------------
// sLSTM (xLSTM sec 2.2) for B=32, T=512, D=1024, U=1024 on gfx1250 (MI455X).
//
// Design (latency-bound serial scan => one persistent kernel):
//   * convert x, W, R to bf16 once (weights become L2-resident: 8MB + 8MB)
//   * persistent kernel: 64 workgroups x 256 threads, one WG per 16-column
//     slice of U. Per timestep each WG computes the 32x16 slice of all four
//     gate blocks: proj = bias + x_t @ W + h @ R via v_wmma_f32_16x16x32_bf16
//     (K-depth 32/instruction, f32 accumulate), then does the stabilized
//     sLSTM pointwise update with c,n,m held in registers across all 512
//     steps. Only h (64KB bf16, double buffered) crosses workgroups, guarded
//     by an agent-scope atomic grid barrier per step.
// ---------------------------------------------------------------------------

#define B_  32
#define T_  512
#define D_  1024
#define U_  1024
#define G_  4096          // 4*U
#define EPSF 1e-8f
#define NWG 64            // U/16 column tiles

typedef __attribute__((ext_vector_type(16))) __bf16 bf16x16;
typedef __attribute__((ext_vector_type(8)))  float  f32x8;
typedef __attribute__((ext_vector_type(4)))  unsigned int u32x4;

union BfPack { u32x4 u[2]; bf16x16 v; };

// Load 16 bf16 (two 16B chunks) into one WMMA operand register block.
__device__ __forceinline__ bf16x16 load_bf16x16(const __bf16* p0, const __bf16* p1) {
    BfPack pk;
    pk.u[0] = *(const u32x4*)p0;
    pk.u[1] = *(const u32x4*)p1;
    return pk.v;
}

// ---------------------------------------------------------------------------
// fp32 -> bf16 conversion
__global__ void cvt_f32_bf16(const float* __restrict__ in, __bf16* __restrict__ out,
                             unsigned n) {
    unsigned i = blockIdx.x * blockDim.x + threadIdx.x;
    if (i < n) out[i] = (__bf16)in[i];
}

// zero h double-buffer + grid-barrier counter (d_ws is poisoned, not zeroed)
__global__ void init_state(__bf16* __restrict__ hbuf, unsigned* __restrict__ counter) {
    unsigned i = blockIdx.x * blockDim.x + threadIdx.x;
    if (i < 2u * B_ * U_) hbuf[i] = (__bf16)0.0f;
    if (i == 0) *counter = 0u;
}

// ---------------------------------------------------------------------------
// device-wide barrier (64 co-resident WGs), monotonically increasing counter
__device__ __forceinline__ void grid_sync(unsigned* counter, unsigned target) {
    __threadfence();            // publish this thread's global stores (agent scope)
    __syncthreads();
    if (threadIdx.x == 0) {
        __hip_atomic_fetch_add(counter, 1u, __ATOMIC_RELEASE, __HIP_MEMORY_SCOPE_AGENT);
        while (__hip_atomic_load(counter, __ATOMIC_ACQUIRE, __HIP_MEMORY_SCOPE_AGENT) < target)
            __builtin_amdgcn_s_sleep(1);
    }
    __syncthreads();
    __threadfence();
}

// stabilized sLSTM pointwise update; c,n,m updated in place (registers)
__device__ __forceinline__ float slstm_point(float ip, float fp, float op, float zp,
                                             float& c, float& n, float& m) {
    float sigf = 1.0f / (1.0f + expf(-fp));
    float logf_ = logf(sigf + EPSF);
    float mt = fmaxf(m + logf_, ip);
    float it = expf(ip - mt);
    float ft = expf(m + logf_ - mt);
    float o  = 1.0f / (1.0f + expf(-op));
    float z  = tanhf(zp);
    c = ft * c + it * z;
    n = ft * n + it;
    m = mt;
    return o * (c / (n + EPSF));
}

// ---------------------------------------------------------------------------
__global__ void __launch_bounds__(256)
slstm_scan(const __bf16* __restrict__ xbf,   // (B,T,D) bf16
           const __bf16* __restrict__ Wbf,   // (D,4U)  bf16
           const __bf16* __restrict__ Rbf,   // (U,4U)  bf16
           const float*  __restrict__ bias,  // (4U,)
           __bf16* __restrict__ hbuf,        // 2 x (B,U) bf16 double buffer
           float*  __restrict__ out,         // (B,T,U) f32
           unsigned* __restrict__ counter) {
    __shared__ float tiles[8][16][16];       // 8 waves x 16x16 f32 gate tiles

    const int wave = threadIdx.x >> 5;
    const int lane = threadIdx.x & 31;
    const int r    = wave >> 2;              // batch row-block (0,1)
    const int g    = wave & 3;               // gate (i,f,o,z)
    const int u0   = blockIdx.x * 16;        // owned U columns [u0, u0+16)
    const int col0 = g * U_ + u0;            // columns in the 4U gate space

    // A-operand lane mapping (16-bit A 16x32, ISA 7.12.2): lane L and L+16
    // both carry row M=L&15; low lanes hold K {0..7,16..23}, high lanes
    // hold K {8..15,24..31}.
    const int mrow  = lane & 15;
    const int brow  = r * 16 + mrow;         // batch index for this lane's A row
    const int khalf = (lane >> 4) * 8;

    const float biasv = bias[col0 + mrow];   // C/D column n = lane&15

    // per-thread sLSTM state for the two pointwise elements this thread owns
    float cs0 = 0.f, ns0 = 0.f, ms0 = 0.f;
    float cs1 = 0.f, ns1 = 0.f, ms1 = 0.f;

    const __bf16* paX = xbf + ((size_t)brow * T_) * D_ + khalf;   // + t*D_ per step
    const size_t  bRowStep = (size_t)32 * G_;                     // B tile K-advance

    for (int t = 0; t < T_; ++t) {
        const __bf16* hin  = hbuf + (size_t)(t & 1) * (B_ * U_);
        __bf16*       hout = hbuf + (size_t)((t + 1) & 1) * (B_ * U_);

        f32x8 acc = { biasv, biasv, biasv, biasv, biasv, biasv, biasv, biasv };

        // ---- proj += x_t @ W  (K = 1024) ----
        const __bf16* pa = paX + (size_t)t * D_;
        const __bf16* pb = Wbf + col0 + mrow;                // B: column n per lane
        #pragma unroll 4
        for (int k0 = 0; k0 < D_; k0 += 32) {
            bf16x16 a  = load_bf16x16(pa + k0, pa + k0 + 16);
            bf16x16 bm; for (int s = 0; s < 16; ++s) bm[s] = pb[(size_t)(khalf + (s & 7) + ((s >> 3) << 4)) * G_];
            __builtin_prefetch(pb + bRowStep, 0, 1);
            pb += bRowStep;
            acc = __builtin_amdgcn_wmma_f32_16x16x32_bf16(
                      false, a, false, bm, (short)0, acc, false, false);
        }

        // ---- proj += h @ R  (K = 1024) ----
        const __bf16* ph = hin + (size_t)brow * U_ + khalf;
        const __bf16* pr = Rbf + col0 + mrow;
        #pragma unroll 4
        for (int k0 = 0; k0 < U_; k0 += 32) {
            bf16x16 a  = load_bf16x16(ph + k0, ph + k0 + 16);
            bf16x16 bm; for (int s = 0; s < 16; ++s) bm[s] = pr[(size_t)(khalf + (s & 7) + ((s >> 3) << 4)) * G_];
            __builtin_prefetch(pr + bRowStep, 0, 1);
            pr += bRowStep;
            acc = __builtin_amdgcn_wmma_f32_16x16x32_bf16(
                      false, a, false, bm, (short)0, acc, false, false);
        }

        // ---- spill gate tile to LDS (C/D layout: m = v + 8*(lane>>4), n = lane&15)
        {
            const int mb = (lane >> 4) * 8;
            const int nn = lane & 15;
            #pragma unroll
            for (int v = 0; v < 8; ++v)
                tiles[wave][mb + v][nn] = acc[v];
        }
        __syncthreads();

        // ---- pointwise sLSTM update: 32x16 elements, 2 per thread ----
        {
            // element 0: e = tid  (batch rows 0..15)
            int e = threadIdx.x;
            int br = e >> 4, nn = e & 15;
            int rb = br >> 4, ml = br & 15;
            float h0 = slstm_point(tiles[rb * 4 + 0][ml][nn],
                                   tiles[rb * 4 + 1][ml][nn],
                                   tiles[rb * 4 + 2][ml][nn],
                                   tiles[rb * 4 + 3][ml][nn], cs0, ns0, ms0);
            int u = u0 + nn;
            out[((size_t)br * T_ + t) * U_ + u] = h0;
            hout[(size_t)br * U_ + u] = (__bf16)h0;

            // element 1: e = tid + 256 (batch rows 16..31)
            e = threadIdx.x + 256;
            br = e >> 4; nn = e & 15;
            rb = br >> 4; ml = br & 15;
            float h1 = slstm_point(tiles[rb * 4 + 0][ml][nn],
                                   tiles[rb * 4 + 1][ml][nn],
                                   tiles[rb * 4 + 2][ml][nn],
                                   tiles[rb * 4 + 3][ml][nn], cs1, ns1, ms1);
            u = u0 + nn;
            out[((size_t)br * T_ + t) * U_ + u] = h1;
            hout[(size_t)br * U_ + u] = (__bf16)h1;
        }

        // publish hout to all WGs; also protects LDS tile reuse next step
        grid_sync(counter, (unsigned)NWG * (unsigned)(t + 1));
    }
}

// ---------------------------------------------------------------------------
extern "C" void kernel_launch(void* const* d_in, const int* in_sizes, int n_in,
                              void* d_out, int out_size, void* d_ws, size_t ws_size,
                              hipStream_t stream) {
    const float* x    = (const float*)d_in[0];   // (B,T,D)
    const float* W    = (const float*)d_in[1];   // (D,4U)
    const float* R    = (const float*)d_in[2];   // (U,4U)
    const float* bias = (const float*)d_in[3];   // (4U,)
    float* out = (float*)d_out;

    char* ws = (char*)d_ws;
    __bf16*   Wbf     = (__bf16*)(ws);                          //  8 MiB
    __bf16*   Rbf     = (__bf16*)(ws + ((size_t)8  << 20));     //  8 MiB
    __bf16*   xbf     = (__bf16*)(ws + ((size_t)16 << 20));     // 32 MiB
    __bf16*   hbuf    = (__bf16*)(ws + ((size_t)48 << 20));     // 128 KiB (2 bufs)
    unsigned* counter = (unsigned*)(ws + ((size_t)48 << 20) + ((size_t)1 << 18));

    const unsigned nW = D_ * G_;        // 4,194,304
    const unsigned nR = U_ * G_;        // 4,194,304
    const unsigned nX = B_ * T_ * D_;   // 16,777,216

    cvt_f32_bf16<<<(nW + 255) / 256, 256, 0, stream>>>(W, Wbf, nW);
    cvt_f32_bf16<<<(nR + 255) / 256, 256, 0, stream>>>(R, Rbf, nR);
    cvt_f32_bf16<<<(nX + 255) / 256, 256, 0, stream>>>(x, xbf, nX);
    init_state<<<(2 * B_ * U_ + 255) / 256, 256, 0, stream>>>(hbuf, counter);

    slstm_scan<<<NWG, 256, 0, stream>>>(xbf, Wbf, Rbf, bias, hbuf, out, counter);
}
